// SplineLayer_65884798321345
// MI455X (gfx1250) — matrix operand order repaired and checked by reading, not compiled, not run on hardware
//
#include <hip/hip_runtime.h>

// SplineLayer on gfx1250 (MI455X): fused one-GEMM via v_wmma_f32_16x16x32_f16.
// Expanded K' = IN_F*32 (16 slope + 16 intercept slots per feature): one input
// feature == one K=32 WMMA step. A = (one-hot*x | one-hot) synthesized in VGPRs
// (per-word compare/select, packed f16 mul); B = weights pre-packed to f16 in d_ws.
// Block: 256 thr = 8 waves: kg (split-K half) x wm (2x16 rows) x wn (4x16 cols);
// per wave, per feature: 2 A-frags x 4 B-frags -> 8 WMMA. Register ping-pong
// (B frags + A frags) hides L2 latency and WMMA<->VALU hazards. Split-K combined
// through LDS; d_out is overwritten (never accumulated - it is poisoned).

typedef __attribute__((ext_vector_type(16))) _Float16 v16h;
typedef __attribute__((ext_vector_type(2)))  _Float16 h2;
typedef __attribute__((ext_vector_type(8)))  float    v8f;

#define IN_F   512
#define OUT_F  512
#define NK     16
#define BATCH  2048
#define BM     64
#define BN     128
#define ICH    32
#define IHALF  (IN_F / 2)

// One-time pre-pass: w16[o][i][0..15]=slopes(f16), w16[o][i][16..31]=intercepts(f16).
__global__ __launch_bounds__(256)
void pack_weights_f16(const float* __restrict__ slopes,
                      const float* __restrict__ intercepts,
                      _Float16* __restrict__ w16)
{
    const int t = blockIdx.x * 256 + threadIdx.x;      // one (o,i) per thread
    const float* ps = slopes     + (size_t)t * NK;
    const float* pi = intercepts + (size_t)t * NK;
    _Float16* pw = w16 + (size_t)t * (2 * NK);
#pragma unroll
    for (int k = 0; k < NK; ++k) {
        pw[k]      = (_Float16)ps[k];
        pw[NK + k] = (_Float16)pi[k];
    }
}

// A-fragment (contiguous order): lane group g owns element e -> K = g*16 + e.
// g==0: halves 0..15 = (e==idx)?x:0 (slope slots),
// g==1: halves 0..15 = (e==idx)?1:0 (intercept slots).
__device__ __forceinline__ v16h make_a_frag(unsigned int xvpk, int idx, int g8)
{
    const unsigned int one = (idx & 1) ? 0x3C000000u : 0x3C00u;   // 1.0h in slot
    const h2 xp = g8 ? (h2)((_Float16)1) : __builtin_bit_cast(h2, xvpk);
    union { unsigned int u[8]; v16h v; } A;
#pragma unroll
    for (int w = 0; w < 8; ++w) {
        const h2 o = __builtin_bit_cast(h2, ((idx >> 1) == w) ? one : 0u);
        A.u[w] = __builtin_bit_cast(unsigned int, (h2)(o * xp));
    }
    return A.v;
}

template <bool PACKED>
__global__ __launch_bounds__(256)
void spline_wmma_kernel(const float* __restrict__ x,
                        const float* __restrict__ slopes,
                        const float* __restrict__ intercepts,
                        const float* __restrict__ bias,
                        const _Float16* __restrict__ w16,
                        float* __restrict__ out)
{
    // 32 KB: staging [kg][il][{xpk, idx}][b]; aliased later as split-K buffer.
    __shared__ unsigned int xinfo[2][ICH][2][BM];

    const int t    = threadIdx.x;
    const int lane = t & 31;
    const int wid  = t >> 5;
    const int kg   = wid & 1;       // split-K half
    const int ws   = wid >> 1;      // 0..3
    const int wm   = ws & 1;
    const int wn   = ws >> 1;       // 0..1
    const int g    = lane >> 4;
    const int g8   = g * 8;
    const int ln   = lane & 15;

    const int b0 = blockIdx.y * BM;
    const int o0 = blockIdx.x * BN;

    int on[4];
#pragma unroll
    for (int nt = 0; nt < 4; ++nt) on[nt] = o0 + (wn * 4 + nt) * 16 + ln;

    const int m0row = wm * 32 + ln;
    const int m1row = wm * 32 + 16 + ln;
    const int io    = kg * IHALF;

    // Loop-invariant per-lane B base pointers (io and g folded in).
    const _Float16* pb[4];
    const float*    pf[4];
#pragma unroll
    for (int nt = 0; nt < 4; ++nt) {
        if (PACKED)
            pb[nt] = w16 + ((size_t)on[nt] * IN_F + io) * (2 * NK) + g * NK;
        else
            pf[nt] = (g ? intercepts : slopes) + ((size_t)on[nt] * IN_F + io) * NK;
    }

    v8f acc[2][4];
#pragma unroll
    for (int mt = 0; mt < 2; ++mt)
#pragma unroll
        for (int nt = 0; nt < 4; ++nt) acc[mt][nt] = (v8f){};

    // ii is the feature index within this split-K half (byte offsets stay small so the
    // compiler can strength-reduce into immediate offsets).
    auto loadB = [&](v16h* d, int ii) {
#pragma unroll
        for (int nt = 0; nt < 4; ++nt) {
            if (PACKED) {
                d[nt] = *(const v16h*)(pb[nt] + ii * (2 * NK));
            } else {
                const float* p = pf[nt] + ii * NK;
                v16h bb;
#pragma unroll
                for (int e = 0; e < NK; ++e) bb[e] = (_Float16)p[e];
                d[nt] = bb;
            }
        }
    };
    auto loadA = [&](v16h* a, int il) {
        const unsigned int xw0 = xinfo[kg][il][0][m0row];
        const unsigned int id0 = xinfo[kg][il][1][m0row];
        const unsigned int xw1 = xinfo[kg][il][0][m1row];
        const unsigned int id1 = xinfo[kg][il][1][m1row];
        a[0] = make_a_frag(xw0, (int)id0, g8);
        a[1] = make_a_frag(xw1, (int)id1, g8);
    };
    auto mma8 = [&](const v16h* a, const v16h* b) {
#pragma unroll
        for (int nt = 0; nt < 4; ++nt) {
            acc[0][nt] = __builtin_amdgcn_wmma_f32_16x16x32_f16(
                false, a[0], false, b[nt], (short)0, acc[0][nt], false, false);
            acc[1][nt] = __builtin_amdgcn_wmma_f32_16x16x32_f16(
                false, a[1], false, b[nt], (short)0, acc[1][nt], false, false);
        }
    };

    for (int i0 = 0; i0 < IHALF; i0 += ICH) {
        __syncthreads();
        {
            // Stage both split-K halves: per (b,i) precompute f16x2(x) + idx.
            const int bl  = t & 63;
            const int q   = t >> 6;          // 0..3
            const int kgs = q >> 1;
            const int igx = (q & 1) * 16;
            const float* src = x + (size_t)(b0 + bl) * IN_F + kgs * IHALF + i0 + igx;
#pragma unroll
            for (int u = 0; u < 16; ++u) {
                const float xv = src[u];
                int idx = (int)(xv * (float)NK);          // x>=0: trunc==floor
                idx = idx < 0 ? 0 : (idx > NK - 1 ? NK - 1 : idx);
                const unsigned short hu =
                    __builtin_bit_cast(unsigned short, (_Float16)xv);
                xinfo[kgs][igx + u][0][bl] = (unsigned int)hu | ((unsigned int)hu << 16);
                xinfo[kgs][igx + u][1][bl] = (unsigned int)idx;
            }
        }
        __syncthreads();

        if (i0 + ICH < IHALF) {   // prefetch next chunk of the weight stream
#pragma unroll
            for (int nt = 0; nt < 4; ++nt) {
                if (PACKED)
                    __builtin_prefetch(pb[nt] + (i0 + ICH) * (2 * NK), 0, 0);
                else
                    __builtin_prefetch(pf[nt] + (i0 + ICH) * NK, 0, 0);
            }
        }

        // Software-pipelined inner loop: B frags and A frags double-buffered.
        v16h bfA[4], bfB[4], aA[2], aB[2];
        loadB(bfA, i0 + 0);
        loadA(aA, 0);
        for (int il = 0; il < ICH; il += 2) {
            loadB(bfB, i0 + il + 1);
            loadA(aB, il + 1);
            mma8(aA, bfA);
            if (il + 2 < ICH) {
                loadB(bfA, i0 + il + 2);
                loadA(aA, il + 2);
            }
            mma8(aB, bfB);
        }
    }

    // ---- split-K combine through LDS (overwrite d_out; no atomics) ----
    __syncthreads();
    float* red = (float*)&xinfo[0][0][0][0];        // 8192 floats = 32 KB
    if (kg == 1) {
        float* dst = red + ws * 2048 + lane * 8;
#pragma unroll
        for (int mt = 0; mt < 2; ++mt)
#pragma unroll
            for (int nt = 0; nt < 4; ++nt) {
                float* d = dst + (mt * 4 + nt) * 256;
#pragma unroll
                for (int r = 0; r < 8; ++r) d[r] = acc[mt][nt][r];
            }
    }
    __syncthreads();
    if (kg == 0) {
        const float* srcr = red + ws * 2048 + lane * 8;
        float bz[4];
#pragma unroll
        for (int nt = 0; nt < 4; ++nt) bz[nt] = bias[on[nt]];
#pragma unroll
        for (int mt = 0; mt < 2; ++mt) {
            const int mb = b0 + wm * 32 + mt * 16 + g8;   // C: M = r + 8*g
#pragma unroll
            for (int nt = 0; nt < 4; ++nt) {
                const float* s = srcr + (mt * 4 + nt) * 256;
#pragma unroll
                for (int r = 0; r < 8; ++r)
                    out[(size_t)(mb + r) * OUT_F + on[nt]] =
                        acc[mt][nt][r] + s[r] + bz[nt];
            }
        }
    }
}

extern "C" void kernel_launch(void* const* d_in, const int* in_sizes, int n_in,
                              void* d_out, int out_size, void* d_ws, size_t ws_size,
                              hipStream_t stream) {
    (void)in_sizes; (void)n_in; (void)out_size;
    const float* x          = (const float*)d_in[0];
    const float* slopes     = (const float*)d_in[1];
    const float* intercepts = (const float*)d_in[2];
    const float* bias       = (const float*)d_in[3];
    float* out              = (float*)d_out;

    const size_t W16_BYTES = (size_t)OUT_F * IN_F * 2 * NK * sizeof(_Float16); // 16.8 MB
    dim3 grid(OUT_F / BN, BATCH / BM);   // 4 x 32 = 128 blocks, 8 waves each

    if (ws_size >= W16_BYTES) {
        _Float16* w16 = (_Float16*)d_ws;
        pack_weights_f16<<<(OUT_F * IN_F) / 256, 256, 0, stream>>>(slopes, intercepts, w16);
        spline_wmma_kernel<true><<<grid, 256, 0, stream>>>(x, slopes, intercepts,
                                                           bias, w16, out);
    } else {
        spline_wmma_kernel<false><<<grid, 256, 0, stream>>>(x, slopes, intercepts,
                                                            bias, nullptr, out);
    }
}
